// GAT_GCN_66391604462263
// MI455X (gfx1250) — matrix operand order repaired and checked by reading, not compiled, not run on hardware
//
#include <hip/hip_runtime.h>
#include <math.h>

// ---------------------------------------------------------------------------
// MI455X / gfx1250 implementation notes
//  - All dense GEMMs use v_wmma_f32_16x16x32_bf16 (wave32 WMMA), f32 accum.
//  - Pipeline is memory-bound (edge gather/scatter ~2 GB @ 23.3 TB/s), so
//    GEMM precision drop to bf16 operands is perf-free and keeps matrix pipes.
//  - GEMM is software-pipelined: global loads of tile k+1 are issued before
//    the fragment loads / WMMAs of tile k (loadcnt overlaps dscnt), and all
//    8 B-fragments are loaded before the WMMA chain so the 8 v_wmma issue
//    back-to-back behind a single s_wait_dscnt.
//  - b1/b2/bg/bf1 biases cancel inside the following batchnorm -> skipped.
// ---------------------------------------------------------------------------

typedef __bf16 bf16x8  __attribute__((ext_vector_type(8)));
typedef __bf16 bf16x16 __attribute__((ext_vector_type(16)));
typedef float  floatx8 __attribute__((ext_vector_type(8)));

__device__ __forceinline__ __bf16 f2bf(float f) {
  unsigned u = __float_as_uint(f);
  u += 0x7FFFu + ((u >> 16) & 1u);               // round-to-nearest-even
  unsigned short s = (unsigned short)(u >> 16);
  return __builtin_bit_cast(__bf16, s);
}

__device__ __forceinline__ void atomicMaxF(float* a, float v) {
  // sign-split trick: works for any mix of pos/neg updates, init = -FLT_MAX
  if (v >= 0.f) atomicMax((int*)a, __float_as_int(v));
  else          atomicMin((unsigned int*)a, __float_as_uint(v));
}

// ---------------------------------------------------------------------------
// Tiled bf16 WMMA GEMM: C[M,N] = A[M,K] * B[K,N]   (A lda==K, C ldc param)
// Block = 256 threads = 8 waves; block tile 128(M) x 128(N), K step 32.
// Wave w owns rows [16w,16w+16) x all 8 N-subtiles -> 8 wmma per K-step/wave.
// ---------------------------------------------------------------------------
__global__ __launch_bounds__(256)
void gemm_bf16_wmma(const float* __restrict__ A, const float* __restrict__ B,
                    float* __restrict__ C, int M, int K, int N, int ldc) {
  __shared__ __align__(16) __bf16 As[128][32];   // 8 KB, row-major [m][k]
  __shared__ __align__(16) __bf16 Bs[128][32];   // 8 KB, transposed [n][k]

  const int tid  = threadIdx.x;
  const int lane = tid & 31;
  const int wave = tid >> 5;
  const int m0 = blockIdx.y * 128;
  const int n0 = blockIdx.x * 128;

  // per-thread staging coordinates (constant across K-steps)
  int aRow[4], aKq[4], bK[4], bNq[4];
  #pragma unroll
  for (int j = 0; j < 4; ++j) {
    int lin = tid + j * 256;
    aRow[j] = lin >> 3;  aKq[j] = (lin & 7) << 2;    // A: 128 rows x 8 float4
    bK[j]   = lin >> 5;  bNq[j] = (lin & 31) << 2;   // B: 32 k-rows x 32 float4
  }

  float4 aReg[4], bReg[4];
  auto load_tiles = [&](int k0) {
    #pragma unroll
    for (int j = 0; j < 4; ++j) {
      int gr = m0 + aRow[j];
      aReg[j] = make_float4(0.f, 0.f, 0.f, 0.f);
      if (gr < M) aReg[j] = *(const float4*)&A[(size_t)gr * K + k0 + aKq[j]];
      bReg[j] = *(const float4*)&B[(size_t)(k0 + bK[j]) * N + n0 + bNq[j]];
    }
  };
  auto stage_tiles = [&]() {
    #pragma unroll
    for (int j = 0; j < 4; ++j) {
      As[aRow[j]][aKq[j] + 0] = f2bf(aReg[j].x);
      As[aRow[j]][aKq[j] + 1] = f2bf(aReg[j].y);
      As[aRow[j]][aKq[j] + 2] = f2bf(aReg[j].z);
      As[aRow[j]][aKq[j] + 3] = f2bf(aReg[j].w);
      Bs[bNq[j] + 0][bK[j]] = f2bf(bReg[j].x);     // transposed: Bs[n][k]
      Bs[bNq[j] + 1][bK[j]] = f2bf(bReg[j].y);
      Bs[bNq[j] + 2][bK[j]] = f2bf(bReg[j].z);
      Bs[bNq[j] + 3][bK[j]] = f2bf(bReg[j].w);
    }
  };

  floatx8 acc[8] = {};
  const int row  = wave * 16 + (lane & 15);
  const int h8   = (lane >> 4) << 3;               // A: lane half -> K+0 / K+8
  const int kh   = (lane >> 4) << 3;               // B: lane half -> K+0 / K+8
  const int coll = lane & 15;

  load_tiles(0);                                   // prologue
  for (int k0 = 0; k0 < K; k0 += 32) {
    stage_tiles();
    __syncthreads();

    // issue next tile's global loads early: loadcnt overlaps the dscnt/WMMA
    if (k0 + 32 < K) load_tiles(k0 + 32);

    // A fragment (16-bit A 16x32 layout)
    bf16x8 alo = *(const bf16x8*)&As[row][h8];
    bf16x8 ahi = *(const bf16x8*)&As[row][16 + h8];
    bf16x16 afrag = __builtin_shufflevector(alo, ahi,
        0,1,2,3,4,5,6,7,8,9,10,11,12,13,14,15);

    // all 8 B fragments first, then 8 back-to-back WMMAs
    bf16x16 bfrag[8];
    #pragma unroll
    for (int nt = 0; nt < 8; ++nt) {
      bf16x8 blo = *(const bf16x8*)&Bs[nt * 16 + coll][kh];
      bf16x8 bhi = *(const bf16x8*)&Bs[nt * 16 + coll][kh + 16];
      bfrag[nt] = __builtin_shufflevector(blo, bhi,
          0,1,2,3,4,5,6,7,8,9,10,11,12,13,14,15);
    }
    #pragma unroll
    for (int nt = 0; nt < 8; ++nt) {
      acc[nt] = __builtin_amdgcn_wmma_f32_16x16x32_bf16(
          false, afrag, false, bfrag[nt], (short)0, acc[nt], false, false);
    }
    __syncthreads();
  }

  // ---- store D (f32 C/D layout: VGPR r -> M = 8*(lane>=16) + r) ----
  const int mloc  = wave * 16 + ((lane >> 4) << 3);
  const int nbase = n0 + (lane & 15);
  #pragma unroll
  for (int nt = 0; nt < 8; ++nt) {
    #pragma unroll
    for (int r = 0; r < 8; ++r) {
      int gr = m0 + mloc + r;
      if (gr < M) C[(size_t)gr * ldc + nbase + nt * 16] = acc[nt][r];
    }
  }
}

// ---------------------------------------------------------------------------
// Attention logits: one wave per (node, head): al_s = <h, a_src>, al_d = <h, a_dst>
// ---------------------------------------------------------------------------
__global__ void attn_logits(const float* __restrict__ h,
                            const float* __restrict__ aS, const float* __restrict__ aD,
                            float* alS, float* alD, int Nn, int H, int C) {
  int wid  = (blockIdx.x * blockDim.x + threadIdx.x) >> 5;
  int lane = threadIdx.x & 31;
  if (wid >= Nn * H) return;
  int n = wid / H, hh = wid - n * H;
  const float* hrow = h + (size_t)n * H * C + (size_t)hh * C;
  const float* as = aS + hh * C;
  const float* ad = aD + hh * C;
  float ss = 0.f, sd = 0.f;
  for (int c = lane; c < C; c += 32) { float v = hrow[c]; ss += v * as[c]; sd += v * ad[c]; }
  for (int o = 16; o; o >>= 1) { ss += __shfl_down(ss, o, 32); sd += __shfl_down(sd, o, 32); }
  if (lane == 0) { alS[wid] = ss; alD[wid] = sd; }
}

__global__ void fill_f32(float* p, float v, int n) {
  int i = blockIdx.x * blockDim.x + threadIdx.x;
  if (i < n) p[i] = v;
}

// edges e<E come from edge_index; e>=E are self loops (src=dst=e-E)
__global__ void edge_lr_max(const int* __restrict__ src, const int* __restrict__ dst,
                            int E, int Etot, const float* __restrict__ alS,
                            const float* __restrict__ alD, float* lrbuf, float* m) {
  int e = blockIdx.x * blockDim.x + threadIdx.x;
  if (e >= Etot) return;
  int s, d;
  if (e < E) { s = src[e]; d = dst[e]; } else { s = d = e - E; }
  #pragma unroll
  for (int h = 0; h < 4; ++h) {
    float v = alS[s * 4 + h] + alD[d * 4 + h];
    float lr = v > 0.f ? v : 0.2f * v;           // leaky_relu 0.2
    lrbuf[(size_t)e * 4 + h] = lr;
    atomicMaxF(&m[d * 4 + h], lr);
  }
}

__global__ void edge_exp_sum(const int* __restrict__ dst, int E, int Etot,
                             const float* __restrict__ m, float* exbuf, float* ssum) {
  int e = blockIdx.x * blockDim.x + threadIdx.x;
  if (e >= Etot) return;
  int d = (e < E) ? dst[e] : e - E;
  #pragma unroll
  for (int h = 0; h < 4; ++h) {
    float ex = expf(exbuf[(size_t)e * 4 + h] - m[d * 4 + h]);
    exbuf[(size_t)e * 4 + h] = ex;
    atomicAdd(&ssum[d * 4 + h], ex);
  }
}

// one wave per edge; lanes sweep HC channels with float4 gathers + atomic scatters
__global__ void gat_agg(const int* __restrict__ src, const int* __restrict__ dst,
                        int E, int Etot, const float* __restrict__ h,
                        const float* __restrict__ exbuf, const float* __restrict__ ssum,
                        float* out, int HC, int logC) {
  int e    = (blockIdx.x * blockDim.x + threadIdx.x) >> 5;
  int lane = threadIdx.x & 31;
  if (e >= Etot) return;
  int s, d;
  if (e < E) { s = src[e]; d = dst[e]; } else { s = d = e - E; }
  float alpha[4];
  #pragma unroll
  for (int hh = 0; hh < 4; ++hh) {
    float den = ssum[d * 4 + hh];
    den = den > 1e-16f ? den : 1e-16f;
    alpha[hh] = exbuf[(size_t)e * 4 + hh] / den;
  }
  const float* hs = h + (size_t)s * HC;
  float* od = out + (size_t)d * HC;
  for (int base = lane * 4; base < HC; base += 128) {
    float4 v = *(const float4*)&hs[base];
    float a = alpha[base >> logC];               // C >= 128 so one head per float4
    atomicAdd(&od[base + 0], v.x * a);
    atomicAdd(&od[base + 1], v.y * a);
    atomicAdd(&od[base + 2], v.z * a);
    atomicAdd(&od[base + 3], v.w * a);
  }
}

__global__ void deg_count(const int* __restrict__ dst, int E, int Etot, float* deg) {
  int e = blockIdx.x * blockDim.x + threadIdx.x;
  if (e >= Etot) return;
  int d = (e < E) ? dst[e] : e - E;
  atomicAdd(&deg[d], 1.f);
}

__global__ void deg_inv(float* deg, int Nn) {
  int n = blockIdx.x * blockDim.x + threadIdx.x;
  if (n < Nn) deg[n] = rsqrtf(fmaxf(deg[n], 1.f));
}

__global__ void gcn_agg(const int* __restrict__ src, const int* __restrict__ dst,
                        int E, int Etot, const float* __restrict__ h3,
                        const float* __restrict__ dinv, float* out) {
  int e    = (blockIdx.x * blockDim.x + threadIdx.x) >> 5;
  int lane = threadIdx.x & 31;
  if (e >= Etot) return;
  int s, d;
  if (e < E) { s = src[e]; d = dst[e]; } else { s = d = e - E; }
  float norm = dinv[s] * dinv[d];
  const float* hs = h3 + (size_t)s * 256;
  float* od = out + (size_t)d * 256;
  for (int base = lane * 4; base < 256; base += 128) {
    float4 v = *(const float4*)&hs[base];
    atomicAdd(&od[base + 0], v.x * norm);
    atomicAdd(&od[base + 1], v.y * norm);
    atomicAdd(&od[base + 2], v.z * norm);
    atomicAdd(&od[base + 3], v.w * norm);
  }
}

// ---------------------------------------------------------------------------
// BatchNorm: pass 1 per-feature sum/sumsq, pass 2 normalize + fused ReLU
// ---------------------------------------------------------------------------
__global__ void bn_stats(const float* __restrict__ X, int rows, int F, float* stats) {
  int f = blockIdx.x * blockDim.x + threadIdx.x;
  if (f >= F) return;
  int per = (rows + gridDim.y - 1) / gridDim.y;
  int r0 = blockIdx.y * per;
  int r1 = min(rows, r0 + per);
  float s = 0.f, q = 0.f;
  for (int r = r0; r < r1; ++r) { float v = X[(size_t)r * F + f]; s += v; q += v * v; }
  atomicAdd(&stats[f], s);
  atomicAdd(&stats[F + f], q);
}

__global__ void bn_norm_relu(float* X, int rows, int F, const float* __restrict__ stats,
                             const float* __restrict__ g, const float* __restrict__ b) {
  int idx = blockIdx.x * blockDim.x + threadIdx.x;
  if (idx >= rows * F) return;
  int f = idx & (F - 1);                         // F is a power of two here
  float inv = 1.f / (float)rows;
  float mu  = stats[f] * inv;
  float var = stats[F + f] * inv - mu * mu;
  float y = (X[idx] - mu) * rsqrtf(var + 1e-5f) * g[f] + b[f];
  X[idx] = y > 0.f ? y : 0.f;
}

// ---------------------------------------------------------------------------
// Pooling + head
// ---------------------------------------------------------------------------
__global__ void pool_sum_max(const float* __restrict__ f3, const int* __restrict__ batch,
                             float* z, int Nn) {
  int idx = blockIdx.x * blockDim.x + threadIdx.x;
  if (idx >= Nn * 256) return;
  int n = idx >> 8, c = idx & 255;
  int g = batch[n];
  float v = f3[idx];
  atomicAdd(&z[(size_t)g * 640 + c], v);
  // f3 >= 0 post-ReLU: uint bit pattern ordering == float ordering, 0-init ok
  atomicMax((unsigned int*)&z[(size_t)g * 640 + 256 + c], __float_as_uint(v));
}

__global__ void cnt_kernel(const int* __restrict__ batch, float* cnt, int Nn) {
  int n = blockIdx.x * blockDim.x + threadIdx.x;
  if (n < Nn) atomicAdd(&cnt[batch[n]], 1.f);
}

__global__ void gap_div(float* z, const float* __restrict__ cnt, int Gg) {
  int idx = blockIdx.x * blockDim.x + threadIdx.x;
  if (idx >= Gg * 256) return;
  int g = idx >> 8, c = idx & 255;
  z[(size_t)g * 640 + c] /= fmaxf(cnt[g], 1.f);
}

__global__ void solv_bias_relu(float* z, const float* __restrict__ bs, int Gg) {
  int idx = blockIdx.x * blockDim.x + threadIdx.x;
  if (idx >= Gg * 128) return;
  int g = idx >> 7, c = idx & 127;
  float* p = &z[(size_t)g * 640 + 512 + c];
  float v = *p + bs[c];
  *p = v > 0.f ? v : 0.f;
}

__global__ void head_out(const float* __restrict__ zf, const float* __restrict__ Wf2,
                         const float* __restrict__ bf2, float* out, int Gg) {
  int g    = (blockIdx.x * blockDim.x + threadIdx.x) >> 5;
  int lane = threadIdx.x & 31;
  if (g >= Gg) return;
  float s = 0.f;
  for (int c = lane; c < 128; c += 32) s += zf[(size_t)g * 128 + c] * Wf2[c];
  for (int o = 16; o; o >>= 1) s += __shfl_down(s, o, 32);
  if (lane == 0) out[g] = 1.f / (1.f + expf(-(s + bf2[0])));
}

// ---------------------------------------------------------------------------
static inline int cdiv(long long a, long long b) { return (int)((a + b - 1) / b); }

extern "C" void kernel_launch(void* const* d_in, const int* in_sizes, int n_in,
                              void* d_out, int out_size, void* d_ws, size_t ws_size,
                              hipStream_t stream) {
  (void)n_in; (void)out_size; (void)ws_size;
  const float* x    = (const float*)d_in[0];
  const int*   ei   = (const int*)d_in[1];
  const int*   batch= (const int*)d_in[2];
  const float* fp   = (const float*)d_in[3];
  const float* W1   = (const float*)d_in[4];
  const float* a1s  = (const float*)d_in[5];
  const float* a1d  = (const float*)d_in[6];
  // d_in[7]=b1, d_in[13]=b2, d_in[17]=bg, d_in[23]=bf1: cancel under batchnorm
  const float* bn1g = (const float*)d_in[8];
  const float* bn1b = (const float*)d_in[9];
  const float* W2   = (const float*)d_in[10];
  const float* a2s  = (const float*)d_in[11];
  const float* a2d  = (const float*)d_in[12];
  const float* bn2g = (const float*)d_in[14];
  const float* bn2b = (const float*)d_in[15];
  const float* Wg   = (const float*)d_in[16];
  const float* bn3g = (const float*)d_in[18];
  const float* bn3b = (const float*)d_in[19];
  const float* Ws   = (const float*)d_in[20];
  const float* bs   = (const float*)d_in[21];
  const float* Wf1  = (const float*)d_in[22];
  const float* bnfg = (const float*)d_in[24];
  const float* bnfb = (const float*)d_in[25];
  const float* Wf2  = (const float*)d_in[26];
  const float* bf2  = (const float*)d_in[27];
  float* out = (float*)d_out;

  const int N = in_sizes[0] / 64;
  const int E = in_sizes[1] / 2;
  const int G = in_sizes[3] / 128;
  const int Etot = E + N;
  const int* src = ei;
  const int* dst = ei + E;

  // workspace layout (f32), total ~420 MB
  float* w = (float*)d_ws;
  float* bufA  = w; w += (size_t)N * 1024;       // GEMM outputs / h3+out3
  float* bufB  = w; w += (size_t)N * 1024;       // aggregated features
  float* alS   = w; w += (size_t)N * 4;
  float* alD   = w; w += (size_t)N * 4;
  float* mb    = w; w += (size_t)N * 4;
  float* sb    = w; w += (size_t)N * 4;
  float* exb   = w; w += (size_t)Etot * 4;
  float* deg   = w; w += N;
  float* stats = w; w += 2048;
  float* z     = w; w += (size_t)G * 640;
  float* zf    = w; w += (size_t)G * 128;
  float* cnt   = w; w += G;

  const int mtiles = cdiv(N, 128);
  const int edgeWaveBlocks = cdiv((long long)Etot * 32, 256);

  // ================= GAT layer 1 : 64 -> 4x128 =================
  gemm_bf16_wmma<<<dim3(4, mtiles), 256, 0, stream>>>(x, W1, bufA, N, 64, 512, 512);
  attn_logits<<<cdiv((long long)N * 4 * 32, 256), 256, 0, stream>>>(bufA, a1s, a1d, alS, alD, N, 4, 128);
  fill_f32<<<cdiv(N * 4, 256), 256, 0, stream>>>(mb, -3.402823466e38f, N * 4);
  hipMemsetAsync(sb, 0, (size_t)N * 4 * sizeof(float), stream);
  edge_lr_max<<<cdiv(Etot, 256), 256, 0, stream>>>(src, dst, E, Etot, alS, alD, exb, mb);
  edge_exp_sum<<<cdiv(Etot, 256), 256, 0, stream>>>(dst, E, Etot, mb, exb, sb);
  hipMemsetAsync(bufB, 0, (size_t)N * 512 * sizeof(float), stream);
  gat_agg<<<edgeWaveBlocks, 256, 0, stream>>>(src, dst, E, Etot, bufA, exb, sb, bufB, 512, 7);
  hipMemsetAsync(stats, 0, 2048 * sizeof(float), stream);
  bn_stats<<<dim3(2, 64), 256, 0, stream>>>(bufB, N, 512, stats);
  bn_norm_relu<<<cdiv((long long)N * 512, 256), 256, 0, stream>>>(bufB, N, 512, stats, bn1g, bn1b);

  // ================= GAT layer 2 : 512 -> 4x256 =================
  gemm_bf16_wmma<<<dim3(8, mtiles), 256, 0, stream>>>(bufB, W2, bufA, N, 512, 1024, 1024);
  attn_logits<<<cdiv((long long)N * 4 * 32, 256), 256, 0, stream>>>(bufA, a2s, a2d, alS, alD, N, 4, 256);
  fill_f32<<<cdiv(N * 4, 256), 256, 0, stream>>>(mb, -3.402823466e38f, N * 4);
  hipMemsetAsync(sb, 0, (size_t)N * 4 * sizeof(float), stream);
  edge_lr_max<<<cdiv(Etot, 256), 256, 0, stream>>>(src, dst, E, Etot, alS, alD, exb, mb);
  edge_exp_sum<<<cdiv(Etot, 256), 256, 0, stream>>>(dst, E, Etot, mb, exb, sb);
  hipMemsetAsync(bufB, 0, (size_t)N * 1024 * sizeof(float), stream);
  gat_agg<<<edgeWaveBlocks, 256, 0, stream>>>(src, dst, E, Etot, bufA, exb, sb, bufB, 1024, 8);
  hipMemsetAsync(stats, 0, 2048 * sizeof(float), stream);
  bn_stats<<<dim3(4, 64), 256, 0, stream>>>(bufB, N, 1024, stats);
  bn_norm_relu<<<cdiv((long long)N * 1024, 256), 256, 0, stream>>>(bufB, N, 1024, stats, bn2g, bn2b);

  // ================= GCN : 1024 -> 256 =================
  float* h3   = bufA;
  float* out3 = bufA + (size_t)N * 256;
  gemm_bf16_wmma<<<dim3(2, mtiles), 256, 0, stream>>>(bufB, Wg, h3, N, 1024, 256, 256);
  hipMemsetAsync(deg, 0, (size_t)N * sizeof(float), stream);
  deg_count<<<cdiv(Etot, 256), 256, 0, stream>>>(dst, E, Etot, deg);
  deg_inv<<<cdiv(N, 256), 256, 0, stream>>>(deg, N);
  hipMemsetAsync(out3, 0, (size_t)N * 256 * sizeof(float), stream);
  gcn_agg<<<edgeWaveBlocks, 256, 0, stream>>>(src, dst, E, Etot, h3, deg, out3);
  hipMemsetAsync(stats, 0, 2048 * sizeof(float), stream);
  bn_stats<<<dim3(1, 64), 256, 0, stream>>>(out3, N, 256, stats);
  bn_norm_relu<<<cdiv((long long)N * 256, 256), 256, 0, stream>>>(out3, N, 256, stats, bn3g, bn3b);

  // ================= Pooling + solvent branch -> z[G,640] =================
  hipMemsetAsync(z, 0, (size_t)G * 640 * sizeof(float), stream);
  hipMemsetAsync(cnt, 0, (size_t)G * sizeof(float), stream);
  pool_sum_max<<<cdiv((long long)N * 256, 256), 256, 0, stream>>>(out3, batch, z, N);
  cnt_kernel<<<cdiv(N, 256), 256, 0, stream>>>(batch, cnt, N);
  gap_div<<<cdiv(G * 256, 256), 256, 0, stream>>>(z, cnt, G);
  // solv = relu(fp @ Ws + bs) written into z[:,512:640] via ldc=640
  gemm_bf16_wmma<<<dim3(1, cdiv(G, 128)), 256, 0, stream>>>(fp, Ws, z + 512, G, 128, 128, 640);
  solv_bias_relu<<<cdiv(G * 128, 256), 256, 0, stream>>>(z, bs, G);

  // ================= MLP head =================
  gemm_bf16_wmma<<<dim3(1, cdiv(G, 128)), 256, 0, stream>>>(z, Wf1, zf, G, 640, 128, 128);
  hipMemsetAsync(stats, 0, 2048 * sizeof(float), stream);
  bn_stats<<<dim3(1, 8), 256, 0, stream>>>(zf, G, 128, stats);
  bn_norm_relu<<<cdiv(G * 128, 256), 256, 0, stream>>>(zf, G, 128, stats, bnfg, bnfb);
  head_out<<<cdiv((long long)G * 32, 256), 256, 0, stream>>>(zf, Wf2, bf2, out, G);
}
